// ByteNetMarkovLM_64914135712145
// MI455X (gfx1250) — matrix operand order repaired and checked by reading, not compiled, not run on hardware
//
#include <hip/hip_runtime.h>
#include <hip/hip_bf16.h>
#include <math.h>

typedef __attribute__((ext_vector_type(16))) _Float16 v16h;
typedef __attribute__((ext_vector_type(8)))  float    v8f;

#define KWIN   9
#define DIM    512
#define LOW    256
#define BATCH  8
#define SEQ    2048
#define WOUT   2040           // SEQ - KWIN + 1
#define NITEMS (BATCH * WOUT) // 16320
#define NI     8              // windows per workgroup
#define NCOLS  72             // NI * KWIN real columns
#define NPAD   80             // padded to 5 N-tiles of 16
#define SD     (DIM + 16)     // 528 f16 column stride (DIM buffers)
#define SL     (LOW + 16)     // 272 f16 column stride (LOW buffers)

// f16 workspace layout (units: f16 elements). Weights pre-swizzled into WMMA
// A-matrix lane order: per tile (16 rows x 32 K), 32 lanes x 16 contiguous f16.
#define A1_LSTRIDE 131072u            // 256x512 per layer: 16*16 tiles * 512
#define A2_OFF     262144u
#define A2_LSTRIDE 327680u            // 5 taps * 128 tiles * 512
#define A2_DKSTR   65536u
#define A3_OFF     917504u
#define A3_LSTRIDE 131072u            // 512x256 per layer: 32*8 tiles * 512

// LDS (f16 units)
#define OFF_X  0
#define OFF_U  (NPAD * SD)            // 42240
#define OFF_L  (2 * NPAD * SD)       // 84480
#define F16_TOT (2 * NPAD * SD + NPAD * SL)   // 106240
#define SMEM_BYTES (F16_TOT * 2 + 72 * 4 + 224 * 4)  // 213664

// K index inside a lane's v16h (contiguous order used here; B loads match):
// lane group g (lane>>4): elements 0..15 hold K = g*16 + 0..15
__device__ __forceinline__ int kmapA(int j, int grp) { return j + grp * 16; }

__device__ __forceinline__ float gelu_exact(float x) {
    return 0.5f * x * (1.0f + erff(x * 0.70710678118654752f));
}

// ---------------------------------------------------------------------------
// Prep: convert f32 conv weights to f16 in WMMA A-tile lane layout.
// 73728 groups of 16 f16; dst offset == gid*16 by construction.
// ---------------------------------------------------------------------------
__global__ __launch_bounds__(256) void prep_weights(
    const float* __restrict__ c1w,   // (2, 256, 512)
    const float* __restrict__ c2w,   // (2, 256, 256, 5)
    const float* __restrict__ c3w,   // (2, 512, 256)
    _Float16* __restrict__ ws)
{
    int gid  = blockIdx.x * 256 + threadIdx.x;
    int lane = gid & 31;
    int grp  = lane >> 4;
    _Float16* dst = ws + (size_t)gid * 16;

    if (gid < 16384) {                       // A1: w1 (256x512), KT=16
        int layer = gid >> 13;
        int tile  = (gid & 8191) >> 5;
        int mt = tile >> 4, kt = tile & 15;
        int row = mt * 16 + (lane & 15);
        const float* src = c1w + (size_t)layer * 131072 + row * 512 + kt * 32;
        #pragma unroll
        for (int j = 0; j < 16; ++j) dst[j] = (_Float16)src[kmapA(j, grp)];
    } else if (gid < 57344) {                // A2: w2[:,:,dk] (256x256), KT=8
        int g2 = gid - 16384;
        int layer = g2 / 20480;
        int r = g2 % 20480;
        int dk = r / 4096;
        int r2 = r % 4096;
        int tile = r2 >> 5;
        int mt = tile >> 3, kt = tile & 7;
        int row = mt * 16 + (lane & 15);
        const float* src = c2w + (size_t)layer * 327680 + row * 1280 + (kt * 32) * 5 + dk;
        #pragma unroll
        for (int j = 0; j < 16; ++j) dst[j] = (_Float16)src[kmapA(j, grp) * 5];
    } else {                                 // A3: w3 (512x256), KT=8
        int g3 = gid - 57344;
        int layer = g3 >> 13;
        int r = g3 & 8191;
        int tile = r >> 5;
        int mt = tile >> 3, kt = tile & 7;
        int row = mt * 16 + (lane & 15);
        const float* src = c3w + (size_t)layer * 131072 + row * 256 + kt * 32;
        #pragma unroll
        for (int j = 0; j < 16; ++j) dst[j] = (_Float16)src[kmapA(j, grp)];
    }
}

// Instance norm over the 9 window positions (one thread owns a full row) + GELU.
template <int C>
__device__ __forceinline__ void norm_gelu(const _Float16* __restrict__ src,
                                          _Float16* __restrict__ dst,
                                          int stride,
                                          const float* __restrict__ g,
                                          const float* __restrict__ b,
                                          int tid)
{
    for (int r = tid; r < C * NI; r += 256) {
        int c  = r & (C - 1);
        int it = r / C;
        const _Float16* p = src + (size_t)(it * KWIN) * stride + c;
        _Float16*       q = dst + (size_t)(it * KWIN) * stride + c;
        float xv[KWIN];
        float m = 0.f;
        #pragma unroll
        for (int k = 0; k < KWIN; ++k) { xv[k] = (float)p[k * stride]; m += xv[k]; }
        m *= (1.0f / 9.0f);
        float v = 0.f;
        #pragma unroll
        for (int k = 0; k < KWIN; ++k) { float d = xv[k] - m; v += d * d; }
        v *= (1.0f / 9.0f);
        float rs = rsqrtf(v + 1e-5f);
        float gg = g[c] * rs, bb = b[c];
        #pragma unroll
        for (int k = 0; k < KWIN; ++k)
            q[k * stride] = (_Float16)gelu_exact((xv[k] - m) * gg + bb);
    }
}

// ---------------------------------------------------------------------------
// Main: one workgroup = 8 windows, full 2-layer ByteNet + head, all in LDS.
// ---------------------------------------------------------------------------
__global__ __launch_bounds__(256) void bytenet_kernel(
    const int*   __restrict__ x,     // (8, 2048)
    const float* __restrict__ emb,   // (7, 512)
    const float* __restrict__ ln1w, const float* __restrict__ ln1b,
    const float* __restrict__ ln2w, const float* __restrict__ ln2b,
    const float* __restrict__ ln3w, const float* __restrict__ ln3b,
    const float* __restrict__ c1b,  const float* __restrict__ c2b,
    const float* __restrict__ c3b,
    const float* __restrict__ outw, const float* __restrict__ outb,
    const _Float16* __restrict__ wsA,
    float* __restrict__ out)
{
    extern __shared__ _Float16 smem[];
    _Float16* bufX = smem + OFF_X;               // DIM residual / x, stride SD
    _Float16* bufU = smem + OFF_U;               // h1 (stride SD) then conv-out/h5 (stride SL)
    _Float16* bufL = smem + OFF_L;               // c1 out / h3, stride SL
    int*   tok   = (int*)(smem + F16_TOT);
    float* headP = (float*)(tok + 72);

    const int tid  = threadIdx.x;
    const int wave = tid >> 5;
    const int lane = tid & 31;
    const int grp  = lane >> 4;
    const int g0   = blockIdx.x * NI;

    // ---- gather tokens + embeddings into bufX (pad columns zeroed) ----
    if (tid < NCOLS) {
        int it = tid / KWIN, k = tid % KWIN;
        int item = g0 + it;
        int b = item / WOUT, w = item - b * WOUT;
        tok[tid] = x[b * SEQ + w + k];
    }
    __syncthreads();
    for (int i = tid; i < NPAD * DIM; i += 256) {
        int col = i >> 9, c = i & (DIM - 1);
        _Float16 v = (_Float16)0.f;
        if (col < NCOLS) v = (_Float16)emb[tok[col] * DIM + c];
        bufX[col * SD + c] = v;
    }
    __syncthreads();

    for (int L = 0; L < 2; ++L) {
        // ---- h1 = gelu(inorm(x)) into bufU (stride SD) ----
        norm_gelu<DIM>(bufX, bufU, SD, ln1w + L * DIM, ln1b + L * DIM, tid);
        __syncthreads();

        // ---- c1: (256x512) x (512 x 80) -> bufL ----
        {
            const _Float16* A = wsA + (size_t)L * A1_LSTRIDE;
            const float* bias = c1b + L * LOW;
            for (int t = wave; t < 80; t += 8) {
                int mt = t / 5, nt = t % 5;
                int colN = nt * 16 + (lane & 15);
                int chb  = grp * 16;
                v8f acc = {};
                #pragma unroll
                for (int kt = 0; kt < 16; ++kt) {
                    v16h av = *(const v16h*)(A + (size_t)(mt * 16 + kt) * 512 + lane * 16);
                    v16h bv = *(const v16h*)(bufU + (size_t)colN * SD + kt * 32 + chb);
                    acc = __builtin_amdgcn_wmma_f32_16x16x32_f16(
                        false, av, false, bv, (short)0, acc, false, false);
                }
                if (colN < NCOLS) {
                    #pragma unroll
                    for (int r = 0; r < 8; ++r) {
                        int row = mt * 16 + grp * 8 + r;
                        bufL[(size_t)colN * SL + row] = (_Float16)(acc[r] + bias[row]);
                    }
                }
            }
        }
        __syncthreads();

        // ---- h3 = gelu(inorm(.)) in place on bufL ----
        norm_gelu<LOW>(bufL, bufL, SL, ln2w + L * LOW, ln2b + L * LOW, tid);
        __syncthreads();

        // ---- conv5 "same": 5 shifted/masked matmuls, bufL -> bufU (stride SL) ----
        {
            const float* bias = c2b + L * LOW;
            for (int t = wave; t < 80; t += 8) {
                int mt = t / 5, nt = t % 5;
                int colN = nt * 16 + (lane & 15);
                int it = colN / KWIN, kk = colN - it * KWIN;
                int chb = grp * 16;
                v8f acc = {};
                for (int dk = 0; dk < 5; ++dk) {
                    int kp = kk + dk - 2;
                    bool ok = (colN < NCOLS) && (kp >= 0) && (kp < KWIN);
                    int kpc = kp < 0 ? 0 : (kp > 8 ? 8 : kp);
                    const _Float16* bbase = bufL + (size_t)(it * KWIN + kpc) * SL + chb;
                    const _Float16* Ab = wsA + A2_OFF + (size_t)L * A2_LSTRIDE
                                             + (size_t)dk * A2_DKSTR;
                    #pragma unroll
                    for (int kt = 0; kt < 8; ++kt) {
                        v16h av = *(const v16h*)(Ab + (size_t)(mt * 8 + kt) * 512 + lane * 16);
                        v16h bv = {};
                        if (ok) bv = *(const v16h*)(bbase + kt * 32);
                        acc = __builtin_amdgcn_wmma_f32_16x16x32_f16(
                            false, av, false, bv, (short)0, acc, false, false);
                    }
                }
                if (colN < NCOLS) {
                    #pragma unroll
                    for (int r = 0; r < 8; ++r) {
                        int row = mt * 16 + grp * 8 + r;
                        bufU[(size_t)colN * SL + row] = (_Float16)(acc[r] + bias[row]);
                    }
                }
            }
        }
        __syncthreads();

        // ---- h5 = gelu(inorm(.)) in place on bufU (stride SL) ----
        norm_gelu<LOW>(bufU, bufU, SL, ln3w + L * LOW, ln3b + L * LOW, tid);
        __syncthreads();

        // ---- c3: (512x256) x (256 x 80), residual via WMMA C-init, -> bufX ----
        {
            const _Float16* A = wsA + A3_OFF + (size_t)L * A3_LSTRIDE;
            const float* bias = c3b + L * DIM;
            for (int t = wave; t < 160; t += 8) {
                int mt = t / 5, nt = t % 5;
                int colN = nt * 16 + (lane & 15);
                int chb = grp * 16;
                v8f acc;
                #pragma unroll
                for (int r = 0; r < 8; ++r) {
                    int row = mt * 16 + grp * 8 + r;
                    acc[r] = (float)bufX[(size_t)colN * SD + row];   // residual
                }
                #pragma unroll
                for (int kt = 0; kt < 8; ++kt) {
                    v16h av = *(const v16h*)(A + (size_t)(mt * 8 + kt) * 512 + lane * 16);
                    v16h bv = *(const v16h*)(bufU + (size_t)colN * SL + kt * 32 + chb);
                    acc = __builtin_amdgcn_wmma_f32_16x16x32_f16(
                        false, av, false, bv, (short)0, acc, false, false);
                }
                if (colN < NCOLS) {
                    #pragma unroll
                    for (int r = 0; r < 8; ++r) {
                        int row = mt * 16 + grp * 8 + r;
                        bufX[(size_t)colN * SD + row] = (_Float16)(acc[r] + bias[row]);
                    }
                }
            }
        }
        __syncthreads();
    }

    // ---- head: pred[item][v] = sum_{c,k} h[c][k] * outw[v][c*9+k] + outb[v] ----
    if (tid < 224) {
        int q = tid & 3, pair = tid >> 2;
        int il = pair / 7, v = pair % 7;
        const _Float16* hb = bufX + (size_t)(il * KWIN) * SD;
        const float* wv = outw + (size_t)v * (DIM * KWIN);
        float acc = 0.f;
        for (int c = q * 128; c < q * 128 + 128; ++c) {
            #pragma unroll
            for (int k = 0; k < KWIN; ++k)
                acc += (float)hb[k * SD + c] * wv[c * KWIN + k];
        }
        headP[tid] = acc;
    }
    __syncthreads();
    if (tid < 56) {
        int il = tid / 7, v = tid % 7;
        float s = headP[tid * 4] + headP[tid * 4 + 1] +
                  headP[tid * 4 + 2] + headP[tid * 4 + 3] + outb[v];
        out[(size_t)(g0 + il) * 7 + v] = s;
    }
}

extern "C" void kernel_launch(void* const* d_in, const int* in_sizes, int n_in,
                              void* d_out, int out_size, void* d_ws, size_t ws_size,
                              hipStream_t stream) {
    const int*   x    = (const int*)  d_in[0];
    const float* emb  = (const float*)d_in[1];
    const float* ln1w = (const float*)d_in[2];
    const float* ln1b = (const float*)d_in[3];
    const float* ln2w = (const float*)d_in[4];
    const float* ln2b = (const float*)d_in[5];
    const float* ln3w = (const float*)d_in[6];
    const float* ln3b = (const float*)d_in[7];
    const float* c1w  = (const float*)d_in[8];
    const float* c1bb = (const float*)d_in[9];
    const float* c2w  = (const float*)d_in[10];
    const float* c2bb = (const float*)d_in[11];
    const float* c3w  = (const float*)d_in[12];
    const float* c3bb = (const float*)d_in[13];
    const float* outw = (const float*)d_in[14];
    const float* outb = (const float*)d_in[15];
    _Float16* wsA = (_Float16*)d_ws;
    float* out = (float*)d_out;

    // swizzle weights to f16 WMMA A-tile layout (2.4MB, L2-resident thereafter)
    prep_weights<<<288, 256, 0, stream>>>(c1w, c2w, c3w, wsA);

    (void)hipFuncSetAttribute((const void*)bytenet_kernel,
                              hipFuncAttributeMaxDynamicSharedMemorySize,
                              (int)SMEM_BYTES);
    bytenet_kernel<<<NITEMS / NI, 256, SMEM_BYTES, stream>>>(
        x, emb, ln1w, ln1b, ln2w, ln2b, ln3w, ln3b,
        c1bb, c2bb, c3bb, outw, outb, wsA, out);
}
